// SuperCKAN_55774445306532
// MI455X (gfx1250) — matrix operand order repaired and checked by reading, not compiled, not run on hardware
//
#include <hip/hip_runtime.h>
#include <hip/hip_bf16.h>

typedef __attribute__((ext_vector_type(16))) _Float16 v16h;
typedef __attribute__((ext_vector_type(8)))  float    v8f;

#define KAN_BATCH 256

// ---------------------------------------------------------------------------
// Closed-form uniform cubic B-spline: at point x only 4 bases are nonzero.
// grid[i] = (i-3)*h - 1, h = 2/GRID; final bases g = 0..GRID+2.
// For cell c = floor((x+1)/h), local u = frac, the nonzero bases are
//   bases[c+0] = (1-u)^3/6
//   bases[c+1] = (3u^3 - 6u^2 + 4)/6
//   bases[c+2] = (-3u^3 + 3u^2 + 3u + 1)/6
//   bases[c+3] = u^3/6
// (indices clamped to [0, GRID+3); out-of-domain x -> no nonzero bases,
// which matches the reference's truncated-grid Cox-de Boor exactly.)
// ---------------------------------------------------------------------------
template<int GRID>
__device__ __forceinline__ int bspline4(float x, float w[4]) {
    const float h = 2.0f / (float)GRID;
    float t  = (x + 1.0f) / h;
    float cf = floorf(t);
    float u  = t - cf;
    float um = 1.0f - u;
    float u2 = u * u;
    float u3 = u2 * u;
    const float s = 1.0f / 6.0f;
    w[0] = um * um * um * s;
    w[1] = (3.0f * u3 - 6.0f * u2 + 4.0f) * s;
    w[2] = (-3.0f * u3 + 3.0f * u2 + 3.0f * u + 1.0f) * s;
    w[3] = u3 * s;
    return (int)cf;
}

// ---------------------------------------------------------------------------
// Fused KAN convolution as one WMMA GEMM.
//   A row (per output position m=(b,c,l)):  [ silu(patch) | bases(patch) ]
//   B (shared, n-major in LDS):             [ base_w ; spline_w * scaler ]
// Each wave computes one 16x16(xNT) output tile; 4 waves per block.
// ---------------------------------------------------------------------------
template<int KSZ, int GRID, int CIN, int NOUT, int HIN, int WIN>
__global__ __launch_bounds__(128)
void kan_conv_wmma(const float* __restrict__ in,        // (B, CIN, HIN, WIN)
                   const float* __restrict__ base_w,    // (NOUT, KK)
                   const float* __restrict__ spline_w,  // (NOUT, KK, G)
                   const float* __restrict__ scaler,    // (NOUT, KK)
                   float* __restrict__ out) {           // (B, CIN*NOUT, HO, WO)
    constexpr int KK = KSZ * KSZ;
    constexpr int G  = GRID + 3;
    constexpr int KR = KK * (G + 1);          // real K (silu + bases)
    constexpr int KP = (KR + 31) & ~31;       // K padded to 32
    constexpr int NP = (NOUT + 15) & ~15;     // N padded to 16
    constexpr int NT = NP / 16;
    constexpr int HO = HIN - KSZ + 1;
    constexpr int WO = WIN - KSZ + 1;
    constexpr int L  = HO * WO;
    constexpr int M  = KAN_BATCH * CIN * L;

    __shared__ __align__(16) _Float16 sB[NP * KP];      // n-major: sB[n*KP + k]
    __shared__ __align__(16) _Float16 sA[4][16 * KP];   // per-wave A tile

    const int tid  = threadIdx.x;
    const int wave = tid >> 5;
    const int lane = tid & 31;
    const int tile = blockIdx.x * 4 + wave;

    // ---- build B = [base_w ; spline_w*scaler]^T (K x N), stored n-major ----
    for (int idx = tid; idx < NP * KP; idx += 128) {
        int n = idx / KP, k = idx % KP;
        float v = 0.0f;
        if (n < NOUT && k < KR) {
            if (k < KK) {
                v = base_w[n * KK + k];
            } else {
                int kk2 = k - KK;
                int ki = kk2 / G, g = kk2 % G;
                v = spline_w[(n * KK + ki) * G + g] * scaler[n * KK + ki];
            }
        }
        sB[idx] = (_Float16)v;
    }

    // ---- zero the whole A tile with b128 stores (covers bases + padding) ----
    {
        float4* p = (float4*)(&sA[wave][0]);            // 16*KP halves = 2*KP f4
        for (int idx = lane; idx < (16 * KP) / 8; idx += 32)
            p[idx] = make_float4(0.0f, 0.0f, 0.0f, 0.0f);
    }

    // ---- build A tile: silu + 4-tap closed-form spline per patch element ----
    for (int t = lane; t < 16 * KK; t += 32) {
        int r  = t / KK, ki = t % KK;
        int m  = tile * 16 + r;
        int mm = m < M ? m : M - 1;           // clamp loads; stores guarded later
        int l  = mm % L;
        int bc = mm / L;                      // b*CIN + c
        int oy = l / WO, ox = l % WO;
        int di = ki / KSZ, dj = ki % KSZ;
        float v = in[(bc * HIN + oy + di) * WIN + ox + dj];
        float s = v / (1.0f + __expf(-v));    // silu
        sA[wave][r * KP + ki] = (_Float16)s;
        float w[4];
        int c = bspline4<GRID>(v, w);
        #pragma unroll
        for (int j = 0; j < 4; ++j) {
            int g = c + j;
            if (g >= 0 && g < G)
                sA[wave][r * KP + KK + ki * G + g] = (_Float16)w[j];
        }
    }

    __syncthreads();

    // ---- WMMA mainloop: D(16x16) += A(16x32) x B(32x16) over K ----
    const int gh = lane >> 4;                 // lane group (K-half select)
    const int nl = lane & 15;                 // row (A) / col (B,D) within tile
    v8f acc[NT] = {};
    for (int kb = 0; kb < KP / 32; ++kb) {
        v16h a;
        #pragma unroll
        for (int e = 0; e < 16; ++e) {        // A layout: K = 16g + e
            int k = kb * 32 + 16 * gh + e;
            a[e] = sA[wave][nl * KP + k];
        }
        #pragma unroll
        for (int nt = 0; nt < NT; ++nt) {
            v16h b;
            #pragma unroll
            for (int e = 0; e < 16; ++e) {    // B layout: K = 16g + e, N = nl
                int k = kb * 32 + 16 * gh + e;
                b[e] = sB[(nt * 16 + nl) * KP + k];
            }
            acc[nt] = __builtin_amdgcn_wmma_f32_16x16x32_f16(
                false, a, false, b, (short)0, acc[nt], false, false);
        }
    }

    // ---- scatter D into (B, CIN*NOUT, HO, WO) ----
    #pragma unroll
    for (int nt = 0; nt < NT; ++nt) {
        int n = nt * 16 + nl;
        if (n < NOUT) {
            #pragma unroll
            for (int r = 0; r < 8; ++r) {     // D layout: M = r + 8g
                int m = tile * 16 + 8 * gh + r;
                if (m < M) {
                    int l  = m % L;
                    int bc = m / L;
                    int bb = bc / CIN, cc = bc % CIN;
                    int oy = l / WO, ox = l % WO;
                    out[((bb * (CIN * NOUT) + cc * NOUT + n) * HO + oy) * WO + ox] = acc[nt][r];
                }
            }
        }
    }
}

// ---------------------------------------------------------------------------
// 2x2 max pool (floor semantics, matches reference crop).
// ---------------------------------------------------------------------------
__global__ void maxpool2_kernel(const float* __restrict__ in, float* __restrict__ out,
                                int C, int H, int W) {
    int Ho = H >> 1, Wo = W >> 1;
    int total = KAN_BATCH * C * Ho * Wo;
    int i = blockIdx.x * blockDim.x + threadIdx.x;
    if (i >= total) return;
    int x = i % Wo; int t = i / Wo;
    int y = t % Ho; t /= Ho;                  // t = b*C + c
    const float* p = in + ((long)t * H + 2 * y) * W + 2 * x;
    out[i] = fmaxf(fmaxf(p[0], p[1]), fmaxf(p[W], p[W + 1]));
}

// ---------------------------------------------------------------------------
// Dense layer via WMMA: out[m][n] = sum_k A[m][k] * Wt[n][k] + bias[n]
// M, N multiples of 16; Nvalid masks padded output columns.
// ---------------------------------------------------------------------------
template<int M, int N, int K>
__global__ __launch_bounds__(128)
void fc_wmma(const float* __restrict__ A, const float* __restrict__ Wt,
             const float* __restrict__ bias, float* __restrict__ out, int Nvalid) {
    constexpr int NTN = N / 16;
    const int wave = threadIdx.x >> 5, lane = threadIdx.x & 31;
    const int tile = blockIdx.x * 4 + wave;
    const int tm = tile / NTN, tn = tile % NTN;
    const int gh = lane >> 4, nl = lane & 15;
    const int row  = tm * 16 + nl;
    const int col  = tn * 16 + nl;
    const int colc = col < Nvalid ? col : Nvalid - 1;   // clamp (no EXEC divergence)
    const float cmask = col < Nvalid ? 1.0f : 0.0f;

    v8f acc = {};
    for (int kb = 0; kb < K / 32; ++kb) {
        v16h a, b;
        #pragma unroll
        for (int e = 0; e < 16; ++e) {
            int k = kb * 32 + 16 * gh + e;
            a[e] = (_Float16)A[row * K + k];
        }
        #pragma unroll
        for (int e = 0; e < 16; ++e) {
            int k = kb * 32 + 16 * gh + e;
            b[e] = (_Float16)(Wt[colc * K + k] * cmask);
        }
        acc = __builtin_amdgcn_wmma_f32_16x16x32_f16(
            false, a, false, b, (short)0, acc, false, false);
    }
    #pragma unroll
    for (int r = 0; r < 8; ++r) {
        int m = tm * 16 + 8 * gh + r;
        int n = tn * 16 + nl;
        if (n < Nvalid) out[m * Nvalid + n] = acc[r] + bias[n];
    }
}

// ---------------------------------------------------------------------------
// Launch: conv1 -> pool -> conv2 -> pool -> conv3 -> pool -> fc1 -> fc2
// Workspace ping-pong: cbuf (conv outs, max 3,538,944 f32), pbuf (pool outs,
// max 884,736 f32), fbuf (fc1 out, 65,536 f32) ~= 18 MB total.
// ---------------------------------------------------------------------------
extern "C" void kernel_launch(void* const* d_in, const int* in_sizes, int n_in,
                              void* d_out, int out_size, void* d_ws, size_t ws_size,
                              hipStream_t stream) {
    const float* x         = (const float*)d_in[0];
    const float* base_w1   = (const float*)d_in[1];
    const float* spline_w1 = (const float*)d_in[2];
    const float* scaler1   = (const float*)d_in[3];
    const float* base_w2   = (const float*)d_in[4];
    const float* spline_w2 = (const float*)d_in[5];
    const float* scaler2   = (const float*)d_in[6];
    const float* base_w3   = (const float*)d_in[7];
    const float* spline_w3 = (const float*)d_in[8];
    const float* scaler3   = (const float*)d_in[9];
    const float* w1        = (const float*)d_in[10];
    const float* b1        = (const float*)d_in[11];
    const float* w2        = (const float*)d_in[12];
    const float* b2        = (const float*)d_in[13];

    float* cbuf = (float*)d_ws;               // 3,538,944 floats max
    float* pbuf = cbuf + 3538944;             //   884,736 floats max
    float* fbuf = pbuf + 884736;              //    65,536 floats
    float* outf = (float*)d_out;              // (256, 10)

    // Layer 1: (256,1,28,28) -> conv k=5 grid5 -> (256,12,24,24) -> pool -> (256,12,12,12)
    kan_conv_wmma<5, 5, 1, 12, 28, 28><<<2304, 128, 0, stream>>>(
        x, base_w1, spline_w1, scaler1, cbuf);            // M=147456 -> 9216 tiles
    maxpool2_kernel<<<(256 * 12 * 12 * 12 + 255) / 256, 256, 0, stream>>>(
        cbuf, pbuf, 12, 24, 24);

    // Layer 2: (256,12,12,12) -> conv k=4 grid10 -> (256,144,9,9) -> pool -> (256,144,4,4)
    kan_conv_wmma<4, 10, 12, 12, 12, 12><<<3888, 128, 0, stream>>>(
        pbuf, base_w2, spline_w2, scaler2, cbuf);         // M=248832 -> 15552 tiles
    maxpool2_kernel<<<(256 * 144 * 4 * 4 + 255) / 256, 256, 0, stream>>>(
        cbuf, pbuf, 144, 9, 9);

    // Layer 3: (256,144,4,4) -> conv k=3 grid10 -> (256,3456,2,2) -> pool -> (256,3456)
    kan_conv_wmma<3, 10, 144, 24, 4, 4><<<2304, 128, 0, stream>>>(
        pbuf, base_w3, spline_w3, scaler3, cbuf);         // M=147456 -> 9216 tiles
    maxpool2_kernel<<<(256 * 3456 + 255) / 256, 256, 0, stream>>>(
        cbuf, pbuf, 3456, 2, 2);

    // FC1: (256,3456) @ (3456,256) + b1 ; FC2: (256,256) @ (256,10) + b2
    fc_wmma<256, 256, 3456><<<64, 128, 0, stream>>>(pbuf, w1, b1, fbuf, 256);
    fc_wmma<256, 16, 256><<<4, 128, 0, stream>>>(fbuf, w2, b2, outf, 10);
}
